// MSGGeneratorLSTM_60378650247661
// MI455X (gfx1250) — matrix operand order repaired and checked by reading, not compiled, not run on hardware
//
#include <hip/hip_runtime.h>
#include <hip/hip_bf16.h>
#include <math.h>

// ---------------------------------------------------------------------------
// Problem constants (match reference)
// ---------------------------------------------------------------------------
#define BATCH  16384
#define HID    512
#define FOURH  2048
#define VOCAB  64
#define TSTEPS 20
#define EOS_IDX 1
#define NKSTEP (HID / 32)   // 16 K-blocks of 32

typedef __bf16 bf16_t;
typedef __attribute__((ext_vector_type(16))) __bf16 v16bf;
typedef __attribute__((ext_vector_type(8)))  float  v8f;

// ---------------------------------------------------------------------------
// WMMA tile loaders (bf16, wave32).  Layouts per CDNA5 ISA.
// ---------------------------------------------------------------------------

// A-matrix 16x32 (MxK): lane L -> row L%16; lanes 0-15 hold K={0..7,16..23},
// lanes 16-31 hold K={8..15,24..31}.  Two 16B loads per lane.
__device__ __forceinline__ v16bf load_a_16x32(const bf16_t* __restrict__ base,
                                              int stride, int lane) {
  const int row  = lane & 15;
  const int koff = (lane & 16) ? 8 : 0;
  const bf16_t* p = base + row * stride + koff;
  union { v16bf v; uint4 q[2]; } u;
  u.q[0] = *reinterpret_cast<const uint4*>(p);        // K = koff..koff+7
  u.q[1] = *reinterpret_cast<const uint4*>(p + 16);   // K = koff+16..koff+23
  return u.v;
}

// B-matrix 32x16 (KxN), supplied as 16 N-major rows of length K (i.e. W rows):
// lane L -> column L%16; lanes 0-15 hold K={0..7,16..23}, lanes 16-31 hold
// K={8..15,24..31}.  Two 16B loads per lane.
__device__ __forceinline__ v16bf load_b_32x16(const bf16_t* __restrict__ base,
                                              int stride, int lane) {
  const int n    = lane & 15;
  const int koff = (lane & 16) ? 8 : 0;
  const bf16_t* p = base + n * stride + koff;
  union { v16bf v; uint4 q[2]; } u;
  u.q[0] = *reinterpret_cast<const uint4*>(p);        // K = koff..koff+7
  u.q[1] = *reinterpret_cast<const uint4*>(p + 16);   // K = koff+16..koff+23
  return u.v;
}

__device__ __forceinline__ v8f wmma_bf16(v16bf a, v16bf b, v8f c) {
  return __builtin_amdgcn_wmma_f32_16x16x32_bf16(
      /*neg_a=*/false, a, /*neg_b=*/false, b,
      /*c_mod=*/(short)0, c, /*reuse_a=*/false, /*reuse_b=*/false);
}

// Branchless fast transcendentals (v_exp_f32 + v_rcp_f32); avoids the ocml
// tanhf expansion with its exec-mask branches in the hot epilogue.
__device__ __forceinline__ float fast_sigmoid(float x) {
  return __builtin_amdgcn_rcpf(1.0f + __expf(-x));
}
__device__ __forceinline__ float fast_tanh(float x) {
  return 1.0f - 2.0f * __builtin_amdgcn_rcpf(1.0f + __expf(2.0f * x));
}

// ---------------------------------------------------------------------------
// Init kernel: bf16 conversions, fused gate bias, state init.  Re-run every
// launch so the whole pipeline is deterministic w.r.t. workspace state.
// ---------------------------------------------------------------------------
__global__ __launch_bounds__(256) void lstm_init_kernel(
    const float* __restrict__ enc_h, const float* __restrict__ enc_c,
    const float* __restrict__ x0,    const float* __restrict__ W_ih,
    const float* __restrict__ b_ih,  const float* __restrict__ b_hh,
    const float* __restrict__ W_hh,  const float* __restrict__ W_out,
    bf16_t* __restrict__ h0, float* __restrict__ c,
    bf16_t* __restrict__ whh, bf16_t* __restrict__ wout,
    float* __restrict__ gbias, float* __restrict__ mask,
    float* __restrict__ logp) {
  const int i = blockIdx.x * blockDim.x + threadIdx.x;
  if (i < BATCH * HID) {                     // recurrent state
    h0[i] = (bf16_t)enc_h[i];
    c[i]  = enc_c[i];
  }
  if (i < FOURH * HID) whh[i] = (bf16_t)W_hh[i];
  if (i < VOCAB * HID) wout[i] = (bf16_t)W_out[i];
  if (i < FOURH) {                           // fold constant input through W_ih
    float s = b_ih[i] + b_hh[i];
    const float* wr = W_ih + (size_t)i * VOCAB;
    #pragma unroll 8
    for (int v = 0; v < VOCAB; ++v) s += x0[v] * wr[v];
    gbias[i] = s;
  }
  if (i < BATCH) mask[i] = 1.0f;
  if (i == 0) *logp = 0.0f;
}

// ---------------------------------------------------------------------------
// Gate GEMM + fused LSTM pointwise.  K loop FULLY unrolled (no backedge ->
// no register-rotation moves) with a textual prefetch distance of one
// K-block: stage i+1's 10 loads are issued before stage i's 4 WMMAs, so each
// WMMA group overlaps an in-flight load clause (partial s_wait_loadcnt).
// Wave tile: 16 rows x 16 hidden-cols, all 4 gates (i,f,g,o) in-register.
// gates[:, g*512 + j] = sum_k h[:,k] * W_hh[g*512+j, k]   (+ gbias)
// ---------------------------------------------------------------------------
__global__ __launch_bounds__(128) void lstm_gate_kernel(
    const bf16_t* __restrict__ h_in,   // [BATCH, HID] bf16
    float*        __restrict__ c_io,   // [BATCH, HID] f32 (in/out, tile-owned)
    bf16_t*       __restrict__ h_out,  // [BATCH, HID] bf16
    const bf16_t* __restrict__ whh,    // [FOURH, HID] bf16
    const float*  __restrict__ gbias)  // [FOURH]
{
  const int lane = threadIdx.x & 31;
  const int wave = threadIdx.x >> 5;
  const int task = blockIdx.x * 4 + wave;      // 32768 wave tasks
  const int jt = task & 31;                    // 512/16 hidden-col tiles
  const int mt = task >> 5;                    // 16384/16 row tiles
  const int m0 = mt * 16;
  const int j0 = jt * 16;

  v8f acc[4];
  #pragma unroll
  for (int g = 0; g < 4; ++g)
    #pragma unroll
    for (int v = 0; v < 8; ++v) acc[g][v] = 0.0f;

  const bf16_t* arow = h_in + (size_t)m0 * HID;
  const bf16_t* brow = whh + (size_t)j0 * HID;  // gate g adds g*HID*HID

  // Double buffer; indices are compile-time constants after full unroll, so
  // SROA turns these arrays into pure SSA registers (no copies).
  v16bf a_buf[2];
  v16bf b_buf[2][4];

  a_buf[0] = load_a_16x32(arow, HID, lane);
  #pragma unroll
  for (int g = 0; g < 4; ++g)
    b_buf[0][g] = load_b_32x16(brow + (size_t)g * HID * HID, HID, lane);

  #pragma unroll
  for (int i = 0; i < NKSTEP; ++i) {
    const int cur = i & 1;
    const int nxt = cur ^ 1;
    if (i + 1 < NKSTEP) {                       // compile-time resolvable
      const int kn = (i + 1) * 32;
      a_buf[nxt] = load_a_16x32(arow + kn, HID, lane);
      #pragma unroll
      for (int g = 0; g < 4; ++g)
        b_buf[nxt][g] =
            load_b_32x16(brow + (size_t)g * HID * HID + kn, HID, lane);
    }
    #pragma unroll
    for (int g = 0; g < 4; ++g)
      acc[g] = wmma_bf16(a_buf[cur], b_buf[cur][g], acc[g]);
  }

  // Epilogue: D layout -> lane L, vgpr v: M = v + (L<16?0:8), N = L%16
  const int n = lane & 15;
  const int j = j0 + n;
  const float gbi = gbias[j];
  const float gbf = gbias[j + HID];
  const float gbg = gbias[j + 2 * HID];
  const float gbo = gbias[j + 3 * HID];
  const int mbase = m0 + ((lane & 16) ? 8 : 0);
  #pragma unroll
  for (int v = 0; v < 8; ++v) {
    const size_t idx = (size_t)(mbase + v) * HID + j;
    const float ig = fast_sigmoid(acc[0][v] + gbi);
    const float fg = fast_sigmoid(acc[1][v] + gbf);
    const float gg = fast_tanh(acc[2][v] + gbg);
    const float og = fast_sigmoid(acc[3][v] + gbo);
    const float cn = fg * c_io[idx] + ig * gg;
    c_io[idx] = cn;
    h_out[idx] = (bf16_t)(og * fast_tanh(cn));
  }
}

// ---------------------------------------------------------------------------
// Output head: logits = h @ W_out^T + b_out; softmax/argmax/one-hot;
// mask + log-prob bookkeeping.  One wave per block, 16 rows per block.
// Same fully-unrolled, prefetch-distance-1 K loop.
// ---------------------------------------------------------------------------
__global__ __launch_bounds__(32) void lstm_out_kernel(
    const bf16_t* __restrict__ h,        // [BATCH, HID] bf16
    const bf16_t* __restrict__ wout,     // [VOCAB, HID] bf16
    const float*  __restrict__ b_out,    // [VOCAB]
    float* __restrict__ message_t,       // [BATCH, VOCAB] (this step's slice)
    float* __restrict__ masks_t,         // [BATCH]        (this step's slice)
    float* __restrict__ mask,            // [BATCH] in/out
    float* __restrict__ logp)            // scalar accumulator
{
  __shared__ float lds[16 * VOCAB];
  const int lane = threadIdx.x;
  const int m0 = blockIdx.x * 16;

  v8f acc[4];
  #pragma unroll
  for (int t = 0; t < 4; ++t)
    #pragma unroll
    for (int v = 0; v < 8; ++v) acc[t][v] = 0.0f;

  const bf16_t* arow = h + (size_t)m0 * HID;

  v16bf a_buf[2];
  v16bf b_buf[2][4];
  a_buf[0] = load_a_16x32(arow, HID, lane);
  #pragma unroll
  for (int t = 0; t < 4; ++t)
    b_buf[0][t] = load_b_32x16(wout + (size_t)(t * 16) * HID, HID, lane);

  #pragma unroll
  for (int i = 0; i < NKSTEP; ++i) {
    const int cur = i & 1;
    const int nxt = cur ^ 1;
    if (i + 1 < NKSTEP) {
      const int kn = (i + 1) * 32;
      a_buf[nxt] = load_a_16x32(arow + kn, HID, lane);
      #pragma unroll
      for (int t = 0; t < 4; ++t)
        b_buf[nxt][t] =
            load_b_32x16(wout + (size_t)(t * 16) * HID + kn, HID, lane);
    }
    #pragma unroll
    for (int t = 0; t < 4; ++t)
      acc[t] = wmma_bf16(a_buf[cur], b_buf[cur][t], acc[t]);
  }

  // Spill logits (+bias) to LDS so each row's 64 logits become contiguous.
  const int n = lane & 15;
  const int moff = (lane & 16) ? 8 : 0;
  #pragma unroll
  for (int t = 0; t < 4; ++t) {
    const float bias = b_out[t * 16 + n];
    #pragma unroll
    for (int v = 0; v < 8; ++v)
      lds[(moff + v) * VOCAB + t * 16 + n] = acc[t][v] + bias;
  }
  __syncthreads();

  // Cooperative zero-fill of message rows (16 rows * 64 cols = 256 float4).
  float4 z; z.x = z.y = z.z = z.w = 0.0f;
  float4* msg4 = reinterpret_cast<float4*>(message_t) + (size_t)blockIdx.x * 256;
  #pragma unroll
  for (int i = 0; i < 8; ++i) msg4[i * 32 + lane] = z;

  if (lane < 16) {
    const int m = m0 + lane;
    const float* row = lds + lane * VOCAB;
    // argmax (first max, matching jnp.argmax) and max value
    float mx = row[0]; int am = 0;
    #pragma unroll 8
    for (int j2 = 1; j2 < VOCAB; ++j2) {
      const float l = row[j2];
      if (l > mx) { mx = l; am = j2; }
    }
    float se = 0.0f;
    #pragma unroll 8
    for (int j2 = 0; j2 < VOCAB; ++j2) se += __expf(row[j2] - mx);
    const float mval = mask[m];
    masks_t[m] = mval;                               // entry mask is emitted
    // log(probs[argmax]) = -log(sum exp(l - max))
    atomicAdd(logp, -__logf(se) * mval);
    mask[m] = (am == EOS_IDX) ? 0.0f : mval;         // mask *= (1 - onehot[EOS])
    message_t[(size_t)m * VOCAB + am] = 1.0f;        // one-hot (after zero-fill)
  }
}

// ---------------------------------------------------------------------------
// Host launcher
// ---------------------------------------------------------------------------
extern "C" void kernel_launch(void* const* d_in, const int* in_sizes, int n_in,
                              void* d_out, int out_size, void* d_ws, size_t ws_size,
                              hipStream_t stream) {
  (void)in_sizes; (void)n_in; (void)out_size; (void)ws_size;
  const float* enc_h = (const float*)d_in[0];
  const float* enc_c = (const float*)d_in[1];
  const float* x0    = (const float*)d_in[2];
  const float* W_ih  = (const float*)d_in[3];
  const float* W_hh  = (const float*)d_in[4];
  const float* b_ih  = (const float*)d_in[5];
  const float* b_hh  = (const float*)d_in[6];
  const float* W_out = (const float*)d_in[7];
  const float* b_out = (const float*)d_in[8];

  // Workspace layout
  char* ws = (char*)d_ws;
  const size_t H_BYTES = (size_t)BATCH * HID * sizeof(bf16_t);   // 16 MB
  bf16_t* hbuf0 = (bf16_t*)(ws);
  bf16_t* hbuf1 = (bf16_t*)(ws + H_BYTES);
  float*  cbuf  = (float*) (ws + 2 * H_BYTES);                   // 32 MB
  bf16_t* whh   = (bf16_t*)(ws + 2 * H_BYTES + (size_t)BATCH * HID * 4);
  bf16_t* wout  = (bf16_t*)((char*)whh + (size_t)FOURH * HID * sizeof(bf16_t));
  float*  gbias = (float*) ((char*)wout + (size_t)VOCAB * HID * sizeof(bf16_t));
  float*  mask  = (float*) ((char*)gbias + (size_t)FOURH * sizeof(float));
  bf16_t* hbuf[2] = {hbuf0, hbuf1};

  // Output layout: message [T,B,V] | masks [T,1,B] | logp [1]
  float* out     = (float*)d_out;
  float* message = out;
  float* masks   = out + (size_t)TSTEPS * BATCH * VOCAB;
  float* logp    = masks + (size_t)TSTEPS * BATCH;

  lstm_init_kernel<<<(BATCH * HID) / 256, 256, 0, stream>>>(
      enc_h, enc_c, x0, W_ih, b_ih, b_hh, W_hh, W_out,
      hbuf0, cbuf, whh, wout, gbias, mask, logp);

  for (int t = 0; t < TSTEPS; ++t) {
    bf16_t* h_in  = hbuf[t & 1];
    bf16_t* h_new = hbuf[(t + 1) & 1];
    // 32768 wave tasks / 4 waves per block
    lstm_gate_kernel<<<8192, 128, 0, stream>>>(h_in, cbuf, h_new, whh, gbias);
    lstm_out_kernel<<<BATCH / 16, 32, 0, stream>>>(
        h_new, wout, b_out,
        message + (size_t)t * BATCH * VOCAB,
        masks + (size_t)t * BATCH,
        mask, logp);
  }
}
